// LinearAttention_20650202759979
// MI455X (gfx1250) — matrix operand order repaired and checked by reading, not compiled, not run on hardware
//
#include <hip/hip_runtime.h>

// ---------------------------------------------------------------------------
// Linear attention for MI455X (gfx1250):
//  - all contractions on v_wmma_f32_16x16x32_bf16 (wave32, 16x16 tiles)
//  - big GEMMs use async global->LDS copies (ASYNCcnt) double-buffered
//    against WMMA, with the B tile shared across all 8 waves of the block.
// ---------------------------------------------------------------------------

#define B_  4
#define S_  4096
#define D_  1024
#define H_  16
#define HD_ 64
#define M_  (B_ * S_)   // 16384 rows in the big GEMMs

#define BM_ 128         // block tile M
#define BN_ 64          // block tile N
#define BK_ 32          // k-step (one WMMA K)
#define ABUF_ (BM_ * BK_)          // 4096 bf16 = 8 KB
#define BBUF_ (BN_ * BK_)          // 2048 bf16 = 4 KB
#define STAGE_ (ABUF_ + BBUF_)     // elems per stage

typedef __attribute__((ext_vector_type(16))) __bf16 v16bf;
typedef __attribute__((ext_vector_type(8)))  __bf16 v8bf;
typedef __attribute__((ext_vector_type(8)))  float  v8f;

typedef unsigned short u16;
typedef unsigned int   u32;

__device__ __forceinline__ u16 f2bf(float f) {
  u32 u = __builtin_bit_cast(u32, f);
  u += 0x7FFFu + ((u >> 16) & 1u);      // round-to-nearest-even
  return (u16)(u >> 16);
}
__device__ __forceinline__ float bf2f(u16 h) {
  return __builtin_bit_cast(float, (u32)h << 16);
}

__device__ __forceinline__ v16bf cat8(v8bf lo, v8bf hi) {
  return __builtin_shufflevector(lo, hi, 0, 1, 2, 3, 4, 5, 6, 7,
                                          8, 9, 10, 11, 12, 13, 14, 15);
}

// A fragment (16x32, M x K): lane holds row m = lane&15.
// half = lane>>4: K elems {half*16..+15} contiguous. p = row base + k0.
__device__ __forceinline__ v16bf frag_a(const u16* p, int half) {
  const v8bf lo = *reinterpret_cast<const v8bf*>(p + half * 16);
  const v8bf hi = *reinterpret_cast<const v8bf*>(p + half * 16 + 8);
  return cat8(lo, hi);
}

// B fragment (32x16, K x N): lane holds column n = lane&15, K = half*16..+15
// contiguous.  Operand stored transposed [N][K]; p includes + half*16.
__device__ __forceinline__ v16bf frag_b(const u16* p) {
  const v8bf lo = *reinterpret_cast<const v8bf*>(p);
  const v8bf hi = *reinterpret_cast<const v8bf*>(p + 8);
  return cat8(lo, hi);
}

__device__ __forceinline__ v8f wmma_bf16(v16bf a, v16bf b, v8f c) {
  // (neg_a, A, neg_b, B, c_mod, C, reuse_a, reuse_b)
  return __builtin_amdgcn_wmma_f32_16x16x32_bf16(false, a, false, b,
                                                 (short)0, c, false, false);
}

// --- CDNA5 async global->LDS (GV mode, 16B per lane), ASYNCcnt-tracked -----
__device__ __forceinline__ void async_cp16(unsigned lds_addr, const u16* gsrc) {
  asm volatile("global_load_async_to_lds_b128 %0, %1, off"
               :: "v"(lds_addr), "v"(gsrc) : "memory");
}
__device__ __forceinline__ void wait_async0() {
  asm volatile("s_wait_asynccnt 0" ::: "memory");
}
__device__ __forceinline__ unsigned lds_u32(const void* p) {
  return (unsigned)(size_t)p;   // generic->LDS: low 32 bits are the LDS offset
}

// ---------------------------------------------------------------------------
// fp32 -> bf16 conversion
// ---------------------------------------------------------------------------
__global__ __launch_bounds__(256) void cvt_f32_bf16(
    const float* __restrict__ in, u16* __restrict__ out, int n) {
  const int i = blockIdx.x * 256 + threadIdx.x;
  if (i < n) out[i] = f2bf(in[i]);
}

// W[K][N] fp32 -> Wt[N][K] bf16 (transpose so B-fragment loads are contiguous)
__global__ __launch_bounds__(256) void cvt_transpose(
    const float* __restrict__ W, u16* __restrict__ Wt, int K, int N) {
  const int i = blockIdx.x * 256 + threadIdx.x;
  if (i >= K * N) return;
  const int n = i % N;
  const int k = i / N;
  Wt[(size_t)n * K + k] = f2bf(W[i]);
}

// ---------------------------------------------------------------------------
// Big GEMM: C[M][N] = A[M][K](bf16) * Bt[N][K]^T(bf16) + bias[N]
// 256 threads = 8 waves.  Block tile 128x64, k-step 32, double-buffered LDS
// filled by global_load_async_to_lds_b128; wave w computes rows 16w x all 64
// cols (1 A-fragment -> 4 WMMAs per k-step, B tile shared by all waves).
// 16B sub-chunks are XOR-swizzled (piece ^ (row&3)) to spread LDS banks.
// ---------------------------------------------------------------------------
template <bool F32OUT>
__global__ __launch_bounds__(256) void gemm_bf16_lds(
    const u16* __restrict__ A, const u16* __restrict__ Bt,
    const float* __restrict__ bias, void* __restrict__ Cout,
    int M, int N, int K) {
  __shared__ u16 smem[2 * STAGE_];   // 24 KB of the 320 KB WGP LDS

  const int t = threadIdx.x;
  const int lane = t & 31;
  const int wave = t >> 5;
  const int l15 = lane & 15;
  const int half = lane >> 4;
  const int gm0 = blockIdx.x * BM_;
  const int gn0 = blockIdx.y * BN_;

  // per-thread staging slots: A needs 512 16B-chunks (2/thread), B needs 256.
  const int ca1 = t + 256;
  const int ra0 = t >> 2,   pa0 = t & 3;
  const int ra1 = ca1 >> 2, pa1 = ca1 & 3;
  const int rb  = t >> 2,   pb  = t & 3;
  const u16* gA0 = A  + (size_t)(gm0 + ra0) * K + pa0 * 8;
  const u16* gA1 = A  + (size_t)(gm0 + ra1) * K + pa1 * 8;
  const u16* gB  = Bt + (size_t)(gn0 + rb)  * K + pb  * 8;
  const int la0 = ra0 * BK_ + (pa0 ^ (ra0 & 3)) * 8;
  const int la1 = ra1 * BK_ + (pa1 ^ (ra1 & 3)) * 8;
  const int lb  = ABUF_ + rb * BK_ + (pb ^ (rb & 3)) * 8;

  v8f acc[4] = {{}, {}, {}, {}};

  const int nk = K / BK_;
  // prologue: stage k-step 0 into buffer 0
  {
    const unsigned base = lds_u32(&smem[0]);
    async_cp16(base + la0 * 2, gA0);
    async_cp16(base + la1 * 2, gA1);
    async_cp16(base + lb * 2,  gB);
  }

  for (int kt = 0; kt < nk; ++kt) {
    wait_async0();        // this wave's copies for buffer kt&1 have landed
    __syncthreads();      // ... and everyone else's too

    if (kt + 1 < nk) {    // stage next k-step into the other buffer
      const int koff = (kt + 1) * BK_;
      const unsigned base = lds_u32(&smem[((kt + 1) & 1) * STAGE_]);
      async_cp16(base + la0 * 2, gA0 + koff);
      async_cp16(base + la1 * 2, gA1 + koff);
      async_cp16(base + lb * 2,  gB  + koff);
    }

    const u16* sa = &smem[(kt & 1) * STAGE_];
    const u16* sb = sa + ABUF_;

    const int am = wave * 16 + l15;
    const v8bf alo = *reinterpret_cast<const v8bf*>(
        &sa[am * BK_ + ((2 * half) ^ (am & 3)) * 8]);
    const v8bf ahi = *reinterpret_cast<const v8bf*>(
        &sa[am * BK_ + ((2 * half + 1) ^ (am & 3)) * 8]);
    const v16bf afrag = cat8(alo, ahi);

#pragma unroll
    for (int tile = 0; tile < 4; ++tile) {
      const int bn = tile * 16 + l15;
      const v8bf blo = *reinterpret_cast<const v8bf*>(
          &sb[bn * BK_ + ((2 * half) ^ (bn & 3)) * 8]);
      const v8bf bhi = *reinterpret_cast<const v8bf*>(
          &sb[bn * BK_ + ((2 * half + 1) ^ (bn & 3)) * 8]);
      acc[tile] = wmma_bf16(afrag, cat8(blo, bhi), acc[tile]);
    }
    __syncthreads();      // all waves done reading buffer kt&1
  }

  const int cmbase = gm0 + wave * 16 + half * 8;
#pragma unroll
  for (int tile = 0; tile < 4; ++tile) {
    const int cn = gn0 + tile * 16 + l15;
    const float bvv = bias[cn];
#pragma unroll
    for (int r = 0; r < 8; ++r) {
      const float v = acc[tile][r] + bvv;
      if (F32OUT) ((float*)Cout)[(size_t)(cmbase + r) * N + cn] = v;
      else        ((u16*)Cout)[(size_t)(cmbase + r) * N + cn] = f2bf(v);
    }
  }
}

// ---------------------------------------------------------------------------
// Per-(b,s,h): softmax over head_dim for Q and K; transpose K and V so the
// sequence-reduction GEMM gets contiguous fragments.
//   Qs : [bh][s][d]   Kt : [bh][d][S]   Vt : [bh][e][S]
// ---------------------------------------------------------------------------
__global__ __launch_bounds__(256) void postproc_qkv(
    const u16* __restrict__ Qr, const u16* __restrict__ Kr,
    const u16* __restrict__ Vr, u16* __restrict__ Qs,
    u16* __restrict__ Kt, u16* __restrict__ Vt) {
  const int tid = blockIdx.x * 256 + threadIdx.x;   // over B*S*H, h fastest
  const int h = tid & (H_ - 1);
  const int s = (tid >> 4) & (S_ - 1);
  const int b = tid >> 16;
  const size_t rbase = (size_t)(b * S_ + s) * D_ + h * HD_;
  const int bh = b * H_ + h;

  float buf[HD_];

  // --- Q softmax -> Qs[bh][s][d]
  float mx = -3.4e38f;
#pragma unroll
  for (int d = 0; d < HD_; ++d) { buf[d] = bf2f(Qr[rbase + d]); mx = fmaxf(mx, buf[d]); }
  float sum = 0.f;
#pragma unroll
  for (int d = 0; d < HD_; ++d) { buf[d] = __expf(buf[d] - mx); sum += buf[d]; }
  float inv = 1.0f / sum;
  u16* q = Qs + ((size_t)bh * S_ + s) * HD_;
#pragma unroll
  for (int d = 0; d < HD_; ++d) q[d] = f2bf(buf[d] * inv);

  // --- K softmax -> Kt[bh][d][s]
  mx = -3.4e38f;
#pragma unroll
  for (int d = 0; d < HD_; ++d) { buf[d] = bf2f(Kr[rbase + d]); mx = fmaxf(mx, buf[d]); }
  sum = 0.f;
#pragma unroll
  for (int d = 0; d < HD_; ++d) { buf[d] = __expf(buf[d] - mx); sum += buf[d]; }
  inv = 1.0f / sum;
#pragma unroll
  for (int d = 0; d < HD_; ++d)
    Kt[((size_t)bh * HD_ + d) * S_ + s] = f2bf(buf[d] * inv);

  // --- V transpose -> Vt[bh][e][s]
#pragma unroll
  for (int e = 0; e < HD_; ++e)
    Vt[((size_t)bh * HD_ + e) * S_ + s] = Vr[rbase + e];
}

// ---------------------------------------------------------------------------
// KVt[bh][e][d] = sum_s V[s][e] * K[s][d]  ((K^T V)^T, stored transposed so
// the next GEMM's B-fragments are contiguous).  One block per (b,h);
// 8 waves x 2 tiles cover the 64x64 output; 128 K-steps over S.
// ---------------------------------------------------------------------------
__global__ __launch_bounds__(256) void kv_reduce(
    const u16* __restrict__ Vt, const u16* __restrict__ Kt,
    u16* __restrict__ KVt) {
  const int bh = blockIdx.x;
  const int lane = threadIdx.x & 31;
  const int wave = threadIdx.x >> 5;
  const int tm = wave >> 1;             // e-tile 0..3
  const int tn0 = (wave & 1) * 2;       // d-tile base 0 or 2
  const int l15 = lane & 15;
  const int half = lane >> 4;

  const u16* vt = Vt + (size_t)bh * HD_ * S_;
  const u16* kt = Kt + (size_t)bh * HD_ * S_;
  const u16* pa  = vt + (size_t)(tm * 16 + l15) * S_;
  const u16* pb0 = kt + (size_t)(tn0 * 16 + l15) * S_ + half * 16;
  const u16* pb1 = kt + (size_t)((tn0 + 1) * 16 + l15) * S_ + half * 16;

  v8f acc0 = {}, acc1 = {};
  for (int s0 = 0; s0 < S_; s0 += 32) {
    if (s0 + 256 < S_) {
      __builtin_prefetch(pa + s0 + 256, 0, 0);
      __builtin_prefetch(pb0 + s0 + 256, 0, 0);
      __builtin_prefetch(pb1 + s0 + 256, 0, 0);
    }
    const v16bf a  = frag_a(pa + s0, half);
    const v16bf b0 = frag_b(pb0 + s0);
    const v16bf b1 = frag_b(pb1 + s0);
    acc0 = wmma_bf16(a, b0, acc0);
    acc1 = wmma_bf16(a, b1, acc1);
  }
  u16* o = KVt + (size_t)bh * HD_ * HD_;
#pragma unroll
  for (int r = 0; r < 8; ++r) {
    const int e = tm * 16 + half * 8 + r;
    o[e * HD_ + tn0 * 16 + l15]       = f2bf(acc0[r]);
    o[e * HD_ + (tn0 + 1) * 16 + l15] = f2bf(acc1[r]);
  }
}

// ---------------------------------------------------------------------------
// attn[b][s][h*64+e] = sum_d Q[bh][s][d] * KV[d][e]  (B-fragment from KVt)
// ---------------------------------------------------------------------------
__global__ __launch_bounds__(256) void apply_qkv(
    const u16* __restrict__ Qs, const u16* __restrict__ KVt,
    u16* __restrict__ attn) {
  const int bh = blockIdx.y;
  const int b = bh >> 4, h = bh & 15;
  const int lane = threadIdx.x & 31;
  const int wave = threadIdx.x >> 5;
  const int m0 = blockIdx.x * 32 + (wave >> 2) * 16;  // s
  const int n0 = (wave & 3) * 16;                     // e
  const int l15 = lane & 15;
  const int half = lane >> 4;

  const u16* q  = Qs  + (size_t)bh * S_ * HD_;
  const u16* kv = KVt + (size_t)bh * HD_ * HD_;
  const u16* pa = q  + (size_t)(m0 + l15) * HD_;
  const u16* pb = kv + (size_t)(n0 + l15) * HD_ + half * 16;

  v8f acc = {};
#pragma unroll
  for (int k0 = 0; k0 < HD_; k0 += 32)
    acc = wmma_bf16(frag_a(pa + k0, half), frag_b(pb + k0), acc);

  const int e = n0 + l15;
#pragma unroll
  for (int r = 0; r < 8; ++r) {
    const int s = m0 + half * 8 + r;
    attn[(size_t)(b * S_ + s) * D_ + h * HD_ + e] = f2bf(acc[r]);
  }
}

// ---------------------------------------------------------------------------
// Host launch
// ---------------------------------------------------------------------------
extern "C" void kernel_launch(void* const* d_in, const int* in_sizes, int n_in,
                              void* d_out, int out_size, void* d_ws,
                              size_t ws_size, hipStream_t stream) {
  (void)in_sizes; (void)n_in; (void)out_size; (void)ws_size;

  const float* x  = (const float*)d_in[0];
  const float* Wq = (const float*)d_in[1];
  const float* bq = (const float*)d_in[2];
  const float* Wk = (const float*)d_in[3];
  const float* bk = (const float*)d_in[4];
  const float* Wv = (const float*)d_in[5];
  const float* bv = (const float*)d_in[6];
  const float* Wo = (const float*)d_in[7];
  const float* bo = (const float*)d_in[8];
  float* out = (float*)d_out;

  char* ws = (char*)d_ws;
  size_t off = 0;
  auto take = [&](size_t bytes) -> char* {
    char* p = ws + off;
    off += (bytes + 255) & ~(size_t)255;
    return p;
  };

  u16* Xb  = (u16*)take((size_t)M_ * D_ * 2);   // x in bf16 (reused as attn)
  u16* Wqt = (u16*)take((size_t)D_ * D_ * 2);
  u16* Wkt = (u16*)take((size_t)D_ * D_ * 2);
  u16* Wvt = (u16*)take((size_t)D_ * D_ * 2);
  u16* Wot = (u16*)take((size_t)D_ * D_ * 2);
  u16* Qr  = (u16*)take((size_t)M_ * D_ * 2);
  u16* Kr  = (u16*)take((size_t)M_ * D_ * 2);
  u16* Vr  = (u16*)take((size_t)M_ * D_ * 2);
  u16* Qs  = (u16*)take((size_t)M_ * D_ * 2);
  u16* Ktp = (u16*)take((size_t)M_ * D_ * 2);
  u16* Vtp = (u16*)take((size_t)M_ * D_ * 2);
  u16* KVt = (u16*)take((size_t)B_ * H_ * HD_ * HD_ * 2);
  u16* attn = Xb;  // Xb is dead after the QKV GEMMs; reuse for attn output

  // 1) convert inputs
  cvt_f32_bf16<<<(M_ * D_) / 256, 256, 0, stream>>>(x, Xb, M_ * D_);
  cvt_transpose<<<(D_ * D_) / 256, 256, 0, stream>>>(Wq, Wqt, D_, D_);
  cvt_transpose<<<(D_ * D_) / 256, 256, 0, stream>>>(Wk, Wkt, D_, D_);
  cvt_transpose<<<(D_ * D_) / 256, 256, 0, stream>>>(Wv, Wvt, D_, D_);
  cvt_transpose<<<(D_ * D_) / 256, 256, 0, stream>>>(Wo, Wot, D_, D_);

  // 2) QKV projections (WMMA + async LDS pipeline)
  dim3 gbig(M_ / BM_, D_ / BN_, 1);
  gemm_bf16_lds<false><<<gbig, 256, 0, stream>>>(Xb, Wqt, bq, Qr, M_, D_, D_);
  gemm_bf16_lds<false><<<gbig, 256, 0, stream>>>(Xb, Wkt, bk, Kr, M_, D_, D_);
  gemm_bf16_lds<false><<<gbig, 256, 0, stream>>>(Xb, Wvt, bv, Vr, M_, D_, D_);

  // 3) softmax feature maps + transposes
  postproc_qkv<<<(B_ * S_ * H_) / 256, 256, 0, stream>>>(Qr, Kr, Vr, Qs, Ktp, Vtp);

  // 4) KV = K^T V per head (WMMA reduction over S)
  kv_reduce<<<B_ * H_, 256, 0, stream>>>(Vtp, Ktp, KVt);

  // 5) out = Q @ KV per head (WMMA)
  apply_qkv<<<dim3(S_ / 32, B_ * H_), 256, 0, stream>>>(Qs, KVt, attn);

  // 6) output projection + bias (WMMA, fp32 out)
  gemm_bf16_lds<true><<<gbig, 256, 0, stream>>>(attn, Wot, bo, out, M_, D_, D_);
}
